// custom_att_1d_12489764897267
// MI455X (gfx1250) — matrix operand order repaired and checked by reading, not compiled, not run on hardware
//
#include <hip/hip_runtime.h>
#include <stdint.h>

typedef __bf16 bf16;
typedef __attribute__((ext_vector_type(16))) __bf16 v16bf;
typedef __attribute__((ext_vector_type(8)))  float  v8f;
typedef __attribute__((ext_vector_type(4)))  unsigned int u32x4;
typedef __attribute__((ext_vector_type(8)))  int i32x8;
typedef __attribute__((ext_vector_type(4)))  int i32x4;

// ---------------- workspace layout (bf16 elements) ----------------
// wq  : [64 out][512 K]  K = c*64 + j ; c<7 -> T_W, c==7 -> T_b
// kw  : [64 out][64 in]  (key_W transposed)
// vw  : [64 out][64 in]
// d1  : [64 out][64 in]
// d2  : [32 out][64 in]
// d3  : [64 out][32 in]
#define WQ_OFF 0
#define KW_OFF 32768
#define VW_OFF 36864
#define D1_OFF 40960
#define D2_OFF 45056
#define D3_OFF 47104
#define WS_ELEMS 49152

__global__ __launch_bounds__(256) void prep_weights(
    const float* __restrict__ key_W, const float* __restrict__ value_W,
    const float* __restrict__ T_W,   const float* __restrict__ T_b,
    const float* __restrict__ d1_W,  const float* __restrict__ d2_W,
    const float* __restrict__ d3_W,  bf16* __restrict__ ws)
{
  int idx = blockIdx.x * blockDim.x + threadIdx.x;
  if (idx >= WS_ELEMS) return;
  float v;
  if (idx < KW_OFF) {                    // wq[o][c*64+j] = T_W[c][o*64+j] (c==7 -> T_b)
    int o = idx >> 9, kk = idx & 511, c = kk >> 6, j = kk & 63;
    v = (c < 7) ? T_W[c * 4096 + o * 64 + j] : T_b[o * 64 + j];
  } else if (idx < VW_OFF) {
    int r = idx - KW_OFF; int o = r >> 6, i = r & 63;
    v = key_W[i * 64 + o];
  } else if (idx < D1_OFF) {
    int r = idx - VW_OFF; int o = r >> 6, i = r & 63;
    v = value_W[i * 64 + o];
  } else if (idx < D2_OFF) {
    int r = idx - D1_OFF; int o = r >> 6, i = r & 63;
    v = d1_W[i * 64 + o];
  } else if (idx < D3_OFF) {
    int r = idx - D2_OFF; int o = r >> 6, i = r & 63;      // o < 32
    v = d2_W[i * 32 + o];
  } else {
    int r = idx - D3_OFF; int o = r >> 5, i = r & 31;      // o < 64, i < 32
    v = d3_W[i * 64 + o];
  }
  ws[idx] = (bf16)v;
}

// ---------------- TDM (Tensor Data Mover) global->LDS ----------------
#if __has_include(<hip/amd_detail/amd_gfx1250_TDM.h>)
#define TDM_6ARG 1
#endif

__device__ __forceinline__ void tdm_load_2d(uint32_t lds_addr, uint64_t gaddr,
                                            uint32_t dim0, uint32_t dim1)
{
  // D# group0: count=1 | lds_addr | global_addr[56:0] | type=2
  u32x4 g0;
  g0[0] = 1u;
  g0[1] = lds_addr;
  g0[2] = (uint32_t)gaddr;
  g0[3] = (uint32_t)(gaddr >> 32) | 0x80000000u;   // type=2 at bits 127:126
  // D# group1: data_size=2B; tensor_dim0/1; tile_dim0/1 = full; stride0=dim0
  i32x8 g1;
  g1[0] = 0x00010000;                                            // data_size=1 (2 bytes)
  g1[1] = (int)((dim0 & 0xFFFFu) << 16);                         // tensor_dim0[15:0] @63:48
  g1[2] = (int)(((dim0 >> 16) & 0xFFFFu) | ((dim1 & 0xFFFFu) << 16)); // dim0 hi | dim1 lo
  g1[3] = (int)(((dim1 >> 16) & 0xFFFFu) | ((dim0 & 0xFFFFu) << 16)); // dim1 hi | tile_dim0
  g1[4] = (int)(dim1 & 0xFFFFu);                                 // tile_dim1 ; tile_dim2=0
  g1[5] = (int)dim0;                                             // tensor_dim0_stride lo
  g1[6] = 0;
  g1[7] = 0;
  i32x4 z4; z4[0]=0; z4[1]=0; z4[2]=0; z4[3]=0;
#ifdef TDM_6ARG
  i32x8 z8; for (int i = 0; i < 8; ++i) z8[i] = 0;
  __builtin_amdgcn_tensor_load_to_lds(g0, g1, z4, z4, z8, 0);
#else
  __builtin_amdgcn_tensor_load_to_lds(g0, g1, z4, z4, 0);
#endif
}

__device__ __forceinline__ uint32_t lds_off(const void* p) {
  return (uint32_t)(uintptr_t)p;   // flat LDS aperture: low 32 bits = LDS byte offset
}

// ---------------- WMMA helpers ----------------
union frag_u { uint32_t u[8]; v16bf v; };

__device__ __forceinline__ v8f zero8() {
  v8f z;
#pragma unroll
  for (int i = 0; i < 8; ++i) z[i] = 0.f;
  return z;
}

__device__ __forceinline__ v8f wmma_bf16(v16bf a, v16bf b, v8f c) {
  return __builtin_amdgcn_wmma_f32_16x16x32_bf16(false, a, false, b, (short)0, c,
                                                 false, false);
}

// A fragment (16x32 bf16) from an f32 LDS row.
// Layout: lanes<16 hold K{0..7,16..23}; lanes>=16 hold K{8..15,24..31}.
__device__ __forceinline__ v16bf make_a(const float* row, int jb, int hi)
{
  const float4* p0 = (const float4*)(row + jb + hi * 8);
  const float4* p1 = (const float4*)(row + jb + 16 + hi * 8);
  float4 a0 = p0[0], a1 = p0[1];
  float4 b0 = p1[0], b1 = p1[1];
  v16bf r;
  r[0]  = (bf16)a0.x; r[1]  = (bf16)a0.y;
  r[2]  = (bf16)a0.z; r[3]  = (bf16)a0.w;
  r[4]  = (bf16)a1.x; r[5]  = (bf16)a1.y;
  r[6]  = (bf16)a1.z; r[7]  = (bf16)a1.w;
  r[8]  = (bf16)b0.x; r[9]  = (bf16)b0.y;
  r[10] = (bf16)b0.z; r[11] = (bf16)b0.w;
  r[12] = (bf16)b1.x; r[13] = (bf16)b1.y;
  r[14] = (bf16)b1.z; r[15] = (bf16)b1.w;
  return r;
}

// Packed bf16 splat of a float scale.
__device__ __forceinline__ uint32_t bf16pk(float s) {
  bf16 h = (bf16)s;
  uint16_t b = __builtin_bit_cast(uint16_t, h);
  return (uint32_t)b * 0x00010001u;
}

// Scale a whole bf16 A-fragment with packed-bf16 VALU (v_pk_mul_bf16).
__device__ __forceinline__ frag_u pk_scale(const frag_u& in, uint32_t pk) {
  frag_u out;
#pragma unroll
  for (int i = 0; i < 8; ++i)
    asm volatile("v_pk_mul_bf16 %0, %1, %2" : "=v"(out.u[i]) : "v"(in.u[i]), "v"(pk));
  return out;
}

// B fragment (32x16 bf16) from an LDS weight row W[n][K...]:
// lanes<16 hold K{0..7,16..23}; lanes>=16 hold K{8..15,24..31} (relative to kb).
__device__ __forceinline__ v16bf load_b(const bf16* row, int kb, int hi)
{
  const uint4* p = (const uint4*)(row + kb + hi * 8);
  union { uint4 q[2]; v16bf v; } u;
  u.q[0] = p[0];
  u.q[1] = p[2];
  return u.v;
}

// ---------------- fused forward kernel ----------------
__global__ __launch_bounds__(128, 1) void fused_forward(
    const float* __restrict__ kv_g,   const float* __restrict__ q_g,
    const float* __restrict__ key_b,  const float* __restrict__ value_b,
    const float* __restrict__ d1_b,   const float* __restrict__ d2_b,
    const float* __restrict__ d3_b,   const float* __restrict__ scale_p,
    const bf16* __restrict__ wsb,     float* __restrict__ out_g)
{
  __shared__ __align__(16) bf16  s_wq[64][512];
  __shared__ __align__(16) bf16  s_kw[64][64];
  __shared__ __align__(16) bf16  s_vw[64][64];
  __shared__ __align__(16) bf16  s_d1[64][64];
  __shared__ __align__(16) bf16  s_d2[32][64];
  __shared__ __align__(16) bf16  s_d3[64][32];
  __shared__ __align__(16) float s_kv [4][16][64];
  __shared__ __align__(16) float s_qs [4][16][8];
  __shared__ __align__(16) float s_att[4][16][64];
  __shared__ __align__(16) float s_v2 [4][16][64];

  const int tid  = threadIdx.x;
  const int w    = tid >> 5;
  const int lane = tid & 31;
  const int m    = lane & 15;
  const int hi   = lane >> 4;
  const size_t base = (size_t)blockIdx.x * 64;

  // Phase 0: wave 0 issues TDM descriptors for all weight tiles.
  if (tid < 32) {
    uint64_t g = (uint64_t)(uintptr_t)wsb;
    tdm_load_2d(lds_off(&s_wq[0][0]), g + 2ull * WQ_OFF, 512, 64);
    tdm_load_2d(lds_off(&s_kw[0][0]), g + 2ull * KW_OFF,  64, 64);
    tdm_load_2d(lds_off(&s_vw[0][0]), g + 2ull * VW_OFF,  64, 64);
    tdm_load_2d(lds_off(&s_d1[0][0]), g + 2ull * D1_OFF,  64, 64);
    tdm_load_2d(lds_off(&s_d2[0][0]), g + 2ull * D2_OFF,  64, 32);
    tdm_load_2d(lds_off(&s_d3[0][0]), g + 2ull * D3_OFF,  32, 64);
    __builtin_amdgcn_s_wait_tensorcnt(0);
  }

  // Phase 1: cooperative token load (coalesced float4).
  {
    const float4* src = (const float4*)(kv_g + base * 64);
    float4* dst = (float4*)&s_kv[0][0][0];
#pragma unroll 2
    for (int i = tid; i < 1024; i += 128) dst[i] = src[i];
  }
  for (int i = tid; i < 512; i += 128) {
    int tok = i >> 3, c = i & 7;
    float v = (c < 7) ? q_g[(base + tok) * 7 + c] : 1.0f;   // channel 7 = T_b path
    s_qs[tok >> 4][tok & 15][c] = v;
  }
  __syncthreads();

  const float* kvrow = &s_kv[w][m][0];
  const float  sc = scale_p[0];
  float kb4[4], vb4[4];
#pragma unroll
  for (int t = 0; t < 4; ++t) { kb4[t] = key_b[t * 16 + m]; vb4[t] = value_b[t * 16 + m]; }

  v8f qacc[4], kacc[4], vacc[4];
#pragma unroll
  for (int t = 0; t < 4; ++t) { qacc[t] = zero8(); kacc[t] = zero8(); vacc[t] = zero8(); }

  // Base bf16 A-fragments of kv (K-chunks 0..31 and 32..63), converted once.
  frag_u a0, a1;
  a0.v = make_a(kvrow, 0, hi);
  a1.v = make_a(kvrow, 32, hi);

  // k = kv @ key_W ; v = kv @ value_W   (K = 64 -> 2 chunks, N = 64 -> 4 tiles)
#pragma unroll
  for (int t = 0; t < 4; ++t) {
    kacc[t] = wmma_bf16(a0.v, load_b(&s_kw[t * 16 + m][0],  0, hi), kacc[t]);
    vacc[t] = wmma_bf16(a0.v, load_b(&s_vw[t * 16 + m][0],  0, hi), vacc[t]);
    kacc[t] = wmma_bf16(a1.v, load_b(&s_kw[t * 16 + m][0], 32, hi), kacc[t]);
    vacc[t] = wmma_bf16(a1.v, load_b(&s_vw[t * 16 + m][0], 32, hi), vacc[t]);
  }

  // q = X_aug @ Wq  (K = 512 -> 8 channels x 2 chunks)
  // A-fragment for channel c = query[c] * base fragment (packed bf16 multiply).
#pragma unroll
  for (int c = 0; c < 8; ++c) {
    uint32_t pk = bf16pk(s_qs[w][m][c]);
    frag_u s0 = pk_scale(a0, pk);
    frag_u s1 = pk_scale(a1, pk);
#pragma unroll
    for (int t = 0; t < 4; ++t) {
      qacc[t] = wmma_bf16(s0.v, load_b(&s_wq[t * 16 + m][0], (c * 2    ) * 32, hi), qacc[t]);
      qacc[t] = wmma_bf16(s1.v, load_b(&s_wq[t * 16 + m][0], (c * 2 + 1) * 32, hi), qacc[t]);
    }
  }

  // att = q * (k + kb) ; vv = (v + vb) * scale    (q,k,v share C/D layout)
#pragma unroll
  for (int t = 0; t < 4; ++t)
#pragma unroll
    for (int e = 0; e < 8; ++e) {
      int row = e + hi * 8;
      s_att[w][row][t * 16 + m] = qacc[t][e] * (kacc[t][e] + kb4[t]);
      s_v2 [w][row][t * 16 + m] = (vacc[t][e] + vb4[t]) * sc;
    }
  __syncthreads();

  // Per-token L2-normalize(att) and softmax(vv); x = att_n * sm(vv) into s_att.
  if (lane < 16) {
    float4* ar4 = (float4*)&s_att[w][lane][0];
    float4* vr4 = (float4*)&s_v2[w][lane][0];
    float ss = 0.f, mx = -3.402823e38f;
#pragma unroll
    for (int j = 0; j < 16; ++j) {
      float4 a = ar4[j], v = vr4[j];
      ss += a.x * a.x + a.y * a.y + a.z * a.z + a.w * a.w;
      mx = fmaxf(mx, fmaxf(fmaxf(v.x, v.y), fmaxf(v.z, v.w)));
    }
    float inv = 1.f / fmaxf(sqrtf(ss), 1e-8f);
    float se = 0.f;
#pragma unroll
    for (int j = 0; j < 16; ++j) {
      float4 v = vr4[j];
      v.x = __expf(v.x - mx); v.y = __expf(v.y - mx);
      v.z = __expf(v.z - mx); v.w = __expf(v.w - mx);
      vr4[j] = v;
      se += v.x + v.y + v.z + v.w;
    }
    float rs = inv / se;                 // combined 1/||att|| * 1/sum(exp)
#pragma unroll
    for (int j = 0; j < 16; ++j) {
      float4 a = ar4[j], v = vr4[j];
      a.x *= v.x * rs; a.y *= v.y * rs;
      a.z *= v.z * rs; a.w *= v.w * rs;
      ar4[j] = a;
    }
  }
  __syncthreads();

  // h1 = relu(x @ d1 + b1)
  {
    v8f hacc[4];
#pragma unroll
    for (int t = 0; t < 4; ++t) hacc[t] = zero8();
    const float* xrow = &s_att[w][m][0];
#pragma unroll
    for (int kk = 0; kk < 2; ++kk) {
      v16bf a = make_a(xrow, kk * 32, hi);
#pragma unroll
      for (int t = 0; t < 4; ++t)
        hacc[t] = wmma_bf16(a, load_b(&s_d1[t * 16 + m][0], kk * 32, hi), hacc[t]);
    }
#pragma unroll
    for (int t = 0; t < 4; ++t) {
      float b = d1_b[t * 16 + m];
#pragma unroll
      for (int e = 0; e < 8; ++e)
        s_v2[w][e + hi * 8][t * 16 + m] = fmaxf(hacc[t][e] + b, 0.f);
    }
  }
  __syncthreads();

  // h2 = relu(h1 @ d2 + b2)   (N = 32 -> 2 tiles)
  {
    v8f gacc[2];
    gacc[0] = zero8(); gacc[1] = zero8();
    const float* hrow = &s_v2[w][m][0];
#pragma unroll
    for (int kk = 0; kk < 2; ++kk) {
      v16bf a = make_a(hrow, kk * 32, hi);
#pragma unroll
      for (int t = 0; t < 2; ++t)
        gacc[t] = wmma_bf16(a, load_b(&s_d2[t * 16 + m][0], kk * 32, hi), gacc[t]);
    }
#pragma unroll
    for (int t = 0; t < 2; ++t) {
      float b = d2_b[t * 16 + m];
#pragma unroll
      for (int e = 0; e < 8; ++e)
        s_att[w][e + hi * 8][t * 16 + m] = fmaxf(gacc[t][e] + b, 0.f);
    }
  }
  __syncthreads();

  // out = h2 @ d3 + b3   (K = 32 -> 1 chunk)
  {
    v8f oacc[4];
#pragma unroll
    for (int t = 0; t < 4; ++t) oacc[t] = zero8();
    const float* grow = &s_att[w][m][0];
    v16bf a = make_a(grow, 0, hi);
#pragma unroll
    for (int t = 0; t < 4; ++t)
      oacc[t] = wmma_bf16(a, load_b(&s_d3[t * 16 + m][0], 0, hi), oacc[t]);
#pragma unroll
    for (int t = 0; t < 4; ++t) {
      float b = d3_b[t * 16 + m];
#pragma unroll
      for (int e = 0; e < 8; ++e)
        s_v2[w][e + hi * 8][t * 16 + m] = oacc[t][e] + b;
    }
  }
  __syncthreads();

  // Coalesced float4 store of 64 tokens x 64 channels.
  {
    float4* dst = (float4*)(out_g + base * 64);
    const float4* src = (const float4*)&s_v2[0][0][0];
#pragma unroll 2
    for (int i = tid; i < 1024; i += 128) dst[i] = src[i];
  }
}

extern "C" void kernel_launch(void* const* d_in, const int* in_sizes, int n_in,
                              void* d_out, int out_size, void* d_ws, size_t ws_size,
                              hipStream_t stream)
{
  const float* kv   = (const float*)d_in[0];
  const float* qin  = (const float*)d_in[1];
  const float* keyW = (const float*)d_in[2];
  const float* keyb = (const float*)d_in[3];
  const float* valW = (const float*)d_in[4];
  const float* valb = (const float*)d_in[5];
  const float* TW   = (const float*)d_in[6];
  const float* Tb   = (const float*)d_in[7];
  const float* d1W  = (const float*)d_in[8];
  const float* d1b  = (const float*)d_in[9];
  const float* d2W  = (const float*)d_in[10];
  const float* d2b  = (const float*)d_in[11];
  const float* d3W  = (const float*)d_in[12];
  const float* d3b  = (const float*)d_in[13];
  const float* scl  = (const float*)d_in[14];
  float* out = (float*)d_out;
  bf16* ws = (bf16*)d_ws;

  hipLaunchKernelGGL(prep_weights, dim3((WS_ELEMS + 255) / 256), dim3(256), 0, stream,
                     keyW, valW, TW, Tb, d1W, d2W, d3W, ws);

  const int ntok = in_sizes[0] / 64;           // N = 131072
  hipLaunchKernelGGL(fused_forward, dim3(ntok / 64), dim3(128), 0, stream,
                     kv, qin, keyb, valb, d1b, d2b, d3b, scl, ws, out);
}
